// ModularAdapterBlock_39960375722758
// MI455X (gfx1250) — matrix operand order repaired and checked by reading, not compiled, not run on hardware
//
#include <hip/hip_runtime.h>
#include <cstddef>
#include <cstdint>

typedef __attribute__((ext_vector_type(16))) _Float16 v16h;
typedef __attribute__((ext_vector_type(8)))  _Float16 h8;
typedef __attribute__((ext_vector_type(8)))  float    v8f;
typedef __attribute__((ext_vector_type(4)))  float    f4;
typedef __attribute__((ext_vector_type(4)))  unsigned int u32x4;
typedef __attribute__((ext_vector_type(8)))  int      i32x8;
typedef __attribute__((ext_vector_type(4)))  int      i32x4;

#define DIMD   768
#define RNK    48
#define KAD    4
#define COLS   192
#define SEQ    4096
#define BATCH  8
#define MTILE  128
#define HSTR   200          // Hs stride in halves (conflict-free b128 reads)

// ---------------- LDS arena (bytes) ----------------
// Phase 1 (GEMM1): B1 double buffer + Hs
// Phase 2 (GEMM2): B2 double buffer overlapping everything (after barrier)
#define B1_COL_BYTES  80                     // 32 halves (64B) + 16B TDM pad
#define B1_BYTES      (COLS * B1_COL_BYTES)  // 15,360
#define SM_B1         0
#define SM_HS         (2 * B1_BYTES)         // 30,720
#define B2_COL_BYTES  432                    // 192 halves (384B) + 3 * 16B pads
#define B2_GRP_COLS   96                     // 6 N-tiles per group
#define B2_BYTES      (B2_GRP_COLS * B2_COL_BYTES)   // 41,472
#define SM_B2         0
#define SMEM_BYTES    (2 * B2_BYTES)         // 82,944

// ---------------- ws layout (bytes) ----------------
#define WS_WDT   0          // f16 [192][768]  (overlaps pool partials pre-conv)
#define WS_WUT   294912     // f16 [768][192]
#define WS_POOL  589824     // f32 [8][768]
#define WS_W     614400     // f32 [8][4]
#define WS_BM    614528     // f32 [8][768]

// -------- pass 1: pooled mean over sequence (two-stage, no atomics) --------
__global__ void pool_partial_k(const float* __restrict__ x, float* __restrict__ part) {
    const int d = blockIdx.y * 256 + threadIdx.x;
    const int b = blockIdx.x;
    const int c = blockIdx.z;
    const float* p = x + ((size_t)(b * SEQ + c * 256)) * DIMD + d;
    float s = 0.f;
    #pragma unroll 4
    for (int i = 0; i < 256; ++i) s += p[(size_t)i * DIMD];
    part[(b * 16 + c) * DIMD + d] = s;
}

__global__ void pool_reduce_k(const float* __restrict__ part, float* __restrict__ pool) {
    const int d = blockIdx.y * 256 + threadIdx.x;
    const int b = blockIdx.x;
    float s = 0.f;
    #pragma unroll
    for (int c = 0; c < 16; ++c) s += part[(b * 16 + c) * DIMD + d];
    pool[b * DIMD + d] = s * (1.0f / SEQ);
}

// -------- router: w = softmax(pool @ Wr + br) --------
__global__ void router_k(const float* __restrict__ pool, const float* __restrict__ Wr,
                         const float* __restrict__ br, float* __restrict__ w) {
    const int b = threadIdx.x;
    if (b >= BATCH) return;
    float acc[KAD] = {0.f, 0.f, 0.f, 0.f};
    const float* pp = pool + b * DIMD;
    for (int d = 0; d < DIMD; ++d) {
        const float pv = pp[d];
        #pragma unroll
        for (int k = 0; k < KAD; ++k) acc[k] += pv * Wr[d * KAD + k];
    }
    float mx = -1e30f;
    #pragma unroll
    for (int k = 0; k < KAD; ++k) { acc[k] += br[k]; mx = fmaxf(mx, acc[k]); }
    float sum = 0.f;
    #pragma unroll
    for (int k = 0; k < KAD; ++k) { acc[k] = __expf(acc[k] - mx); sum += acc[k]; }
    const float inv = 1.0f / sum;
    #pragma unroll
    for (int k = 0; k < KAD; ++k) w[b * KAD + k] = acc[k] * inv;
}

// -------- bias_mix[b][d] = sum_k w[b,k] * bu[k,d] --------
__global__ void biasmix_k(const float* __restrict__ w, const float* __restrict__ bu,
                          float* __restrict__ bm) {
    for (int idx = threadIdx.x; idx < BATCH * DIMD; idx += 256) {
        const int b = idx / DIMD, d = idx % DIMD;
        float s = 0.f;
        #pragma unroll
        for (int k = 0; k < KAD; ++k) s += w[b * KAD + k] * bu[k * DIMD + d];
        bm[idx] = s;
    }
}

// -------- weight convert/transpose to f16, K-contiguous columns --------
__global__ void conv_wd_k(const float* __restrict__ Wd, _Float16* __restrict__ WdT) {
    const int g = blockIdx.x * 256 + threadIdx.x;
    const int col = g / DIMD, d = g % DIMD;
    const int k = col / RNK, r = col % RNK;
    WdT[(size_t)col * DIMD + d] = (_Float16)Wd[((size_t)k * DIMD + d) * RNK + r];
}

__global__ void conv_wu_k(const float* __restrict__ Wu, _Float16* __restrict__ WuT) {
    const int g = blockIdx.x * 256 + threadIdx.x;
    const int dout = g / COLS, kk = g % COLS;
    const int k = kk / RNK, r = kk % RNK;
    WuT[(size_t)dout * COLS + kk] = (_Float16)Wu[((size_t)k * RNK + r) * DIMD + dout];
}

__device__ __forceinline__ v8f wmma_f16(v16h a, v16h b, v8f c) {
    return __builtin_amdgcn_wmma_f32_16x16x32_f16(false, a, false, b, (short)0, c, false, false);
}

// -------- TDM: 2D tile load -> LDS with per-row padding --------
// D# per CDNA5 ISA. data_size = 2 bytes (f16).
// pad_interval_f / pad_amount_f are the raw D# field values.
// 6-arg builtin form (this toolchain): (u32x4, i32x8, i32x4, i32x4, i32x8, i32).
__device__ __forceinline__ void tdm_load_2d(
    unsigned lds_addr, const void* gptr,
    unsigned tile_d0, unsigned tile_d1,
    unsigned tensor_d0, unsigned tensor_d1,
    unsigned stride0_elems,
    unsigned pad_interval_f, unsigned pad_amount_f)
{
    const unsigned long long ga = (unsigned long long)(size_t)gptr;
    u32x4 g0;
    g0[0] = 1u;                                              // count=1, user mode
    g0[1] = lds_addr;                                        // LDS byte addr
    g0[2] = (unsigned)(ga & 0xFFFFFFFFu);                    // global_addr[31:0]
    g0[3] = (unsigned)((ga >> 32) & 0x01FFFFFFu) | 0x80000000u; // addr[56:32] | type=2
    i32x8 g1;
    g1[0] = (int)((1u << 16)                                 // data_size = 2B
                | (1u << 20)                                 // pad_enable
                | (pad_interval_f << 22)
                | (pad_amount_f << 25));
    g1[1] = (int)((tensor_d0 & 0xFFFFu) << 16);              // atomic_addr=0 | dim0.lo16
    g1[2] = (int)(((tensor_d0 >> 16) & 0xFFFFu) | ((tensor_d1 & 0xFFFFu) << 16));
    g1[3] = (int)(((tensor_d1 >> 16) & 0xFFFFu) | ((tile_d0 & 0xFFFFu) << 16));
    g1[4] = (int)(tile_d1 & 0xFFFFu);                        // tile_dim2 = 0
    g1[5] = (int)stride0_elems;                              // dim0_stride[31:0]
    g1[6] = 0;                                               // stride hi / dim1_stride.lo
    g1[7] = 0;
    const i32x4 z4 = {0, 0, 0, 0};
    const i32x8 z8 = {0, 0, 0, 0, 0, 0, 0, 0};
    __builtin_amdgcn_tensor_load_to_lds(g0, g1, z4, z4, z8, 0);
}

// -------- main fused kernel --------
__global__ __launch_bounds__(256) void adapter_main_k(
    const float* __restrict__ x, const float* __restrict__ w,
    const float* __restrict__ bd, const float* __restrict__ bm,
    const _Float16* __restrict__ WdT, const _Float16* __restrict__ WuT,
    float* __restrict__ out)
{
    __shared__ __align__(128) unsigned char smem[SMEM_BYTES];

    const int tid  = threadIdx.x;
    const int wave = tid >> 5;
    const int lane = tid & 31;
    const int l15  = lane & 15;
    const int hi   = (lane >> 4) & 1;

    const int tokBase = blockIdx.x * MTILE;
    const int b       = tokBase >> 12;
    const int arow    = wave * 16 + l15;
    const size_t xrow = (size_t)(tokBase + arow) * DIMD;

    const unsigned ldsBase = (unsigned)(size_t)(&smem[0]);   // LDS aperture: low 32 = offset

    float wk[KAD];
    #pragma unroll
    for (int k = 0; k < KAD; ++k) wk[k] = w[b * KAD + k];

    // ================= Phase 1: GEMM1  H = gelu(X @ WdT^T + bd) * w_k =================
    v8f acc[12];
    #pragma unroll
    for (int nt = 0; nt < 12; ++nt) acc[nt] = (v8f){0.f,0.f,0.f,0.f,0.f,0.f,0.f,0.f};

    // prologue: TDM-stage B panel for d0=0 into buffer 0
    if (wave == 0) {
        tdm_load_2d(ldsBase + SM_B1, WdT, /*tile*/32, COLS,
                    /*tensor*/DIMD, COLS, /*stride*/DIMD,
                    /*interval 16 DW*/3, /*amount 4 DW*/3);
        __builtin_amdgcn_s_wait_tensorcnt(0);
    }
    __syncthreads();

    for (int step = 0; step < 24; ++step) {
        const int cur = step & 1;
        // async-stage next K-step's B panel while computing this one
        if (wave == 0 && step < 23) {
            tdm_load_2d(ldsBase + SM_B1 + (cur ^ 1) * B1_BYTES, WdT + (step + 1) * 32,
                        32, COLS, DIMD, COLS, DIMD, 3, 3);
        }
        const int d0 = step * 32;
        // A fragment straight from global f32 (ISA A layout), converted to f16
        const float* ap = x + xrow + d0 + (hi ? 8 : 0);
        const f4 a0 = *(const f4*)(ap);
        const f4 a1 = *(const f4*)(ap + 4);
        const f4 a2 = *(const f4*)(ap + 16);
        const f4 a3 = *(const f4*)(ap + 20);
        v16h A;
        #pragma unroll
        for (int i = 0; i < 4; ++i) {
            A[i]      = (_Float16)a0[i];
            A[4 + i]  = (_Float16)a1[i];
            A[8 + i]  = (_Float16)a2[i];
            A[12 + i] = (_Float16)a3[i];
        }
        const unsigned char* b1 = smem + SM_B1 + cur * B1_BYTES + (hi ? 16 : 0);
        #pragma unroll
        for (int half = 0; half < 2; ++half) {
            v16h Bf[6];
            #pragma unroll
            for (int j = 0; j < 6; ++j) {                    // batch DS loads
                const unsigned char* cb = b1 + ((half * 6 + j) * 16 + l15) * B1_COL_BYTES;
                const h8 blo = *(const h8*)(cb);
                const h8 bhi = *(const h8*)(cb + 32);
                Bf[j] = __builtin_shufflevector(blo, bhi,
                    0,1,2,3,4,5,6,7,8,9,10,11,12,13,14,15);
            }
            #pragma unroll
            for (int j = 0; j < 6; ++j)
                acc[half * 6 + j] = wmma_f16(A, Bf[j], acc[half * 6 + j]);
        }
        if (wave == 0) __builtin_amdgcn_s_wait_tensorcnt(0);
        __syncthreads();
    }

    // epilogue: +bd, gelu(tanh), * router weight -> Hs (in-wave transpose via LDS)
    _Float16* Hs = (_Float16*)(smem + SM_HS);
    #pragma unroll
    for (int nt = 0; nt < 12; ++nt) {
        const int col  = nt * 16 + l15;
        const float bdv = bd[col];
        const float wkk = wk[nt / 3];
        #pragma unroll
        for (int v = 0; v < 8; ++v) {
            float hv = acc[nt][v] + bdv;
            const float u = 0.7978845608028654f * (hv + 0.044715f * hv * hv * hv);
            const float g = 0.5f * hv * (1.0f + tanhf(u));
            Hs[(wave * 16 + v + hi * 8) * HSTR + col] = (_Float16)(g * wkk);
        }
    }
    // per-wave producer/consumer on own rows: DScnt ordering suffices, no barrier yet
    v16h Af[6];
    #pragma unroll
    for (int s = 0; s < 6; ++s) {
        const _Float16* hp = &Hs[(wave * 16 + l15) * HSTR + s * 32 + (hi ? 16 : 0)];
        const h8 alo = *(const h8*)hp;
        const h8 ahi = *(const h8*)(hp + 8);
        Af[s] = __builtin_shufflevector(alo, ahi,
            0,1,2,3,4,5,6,7,8,9,10,11,12,13,14,15);
    }
    __syncthreads();   // everyone done with Hs; B2 buffers may now overwrite it

    // ================= Phase 2: GEMM2  out = H @ WuT^T + bias_mix =================
    if (wave == 0) {
        tdm_load_2d(ldsBase + SM_B2, WuT, /*tile*/COLS, B2_GRP_COLS,
                    COLS, DIMD, COLS, /*interval 32 DW*/4, /*amount 4 DW*/3);
        __builtin_amdgcn_s_wait_tensorcnt(0);
    }
    __syncthreads();

    for (int g = 0; g < 8; ++g) {                            // 8 groups of 6 N-tiles
        const int cur = g & 1;
        if (wave == 0 && g < 7) {
            tdm_load_2d(ldsBase + SM_B2 + (cur ^ 1) * B2_BYTES,
                        WuT + (size_t)(g + 1) * B2_GRP_COLS * COLS,
                        COLS, B2_GRP_COLS, COLS, DIMD, COLS, 4, 3);
        }
        const unsigned char* b2 = smem + SM_B2 + cur * B2_BYTES;
        #pragma unroll
        for (int ntloc = 0; ntloc < 6; ++ntloc) {
            const int nt  = g * 6 + ntloc;
            const int col = nt * 16 + l15;
            v16h Bf[6];
            #pragma unroll
            for (int s = 0; s < 6; ++s) {                    // batch DS loads
                const int kdw = s * 16 + (hi ? 8 : 0);       // DW offset in column
                const unsigned char* cb = b2 + (ntloc * 16 + l15) * B2_COL_BYTES
                                        + 4 * kdw + ((kdw >> 5) << 4);  // skip pads
                const h8 blo = *(const h8*)(cb);
                const h8 bhi = *(const h8*)(cb + 16);
                Bf[s] = __builtin_shufflevector(blo, bhi,
                    0,1,2,3,4,5,6,7,8,9,10,11,12,13,14,15);
            }
            v8f c = (v8f){0.f,0.f,0.f,0.f,0.f,0.f,0.f,0.f};
            #pragma unroll
            for (int s = 0; s < 6; ++s) c = wmma_f16(Af[s], Bf[s], c);
            const float bmv = bm[b * DIMD + col];
            #pragma unroll
            for (int v = 0; v < 8; ++v)
                out[(size_t)(tokBase + wave * 16 + v + hi * 8) * DIMD + col] = c[v] + bmv;
        }
        if (wave == 0) __builtin_amdgcn_s_wait_tensorcnt(0);
        __syncthreads();
    }
}

extern "C" void kernel_launch(void* const* d_in, const int* in_sizes, int n_in,
                              void* d_out, int out_size, void* d_ws, size_t ws_size,
                              hipStream_t stream) {
    (void)in_sizes; (void)n_in; (void)out_size; (void)ws_size;
    const float* x  = (const float*)d_in[0];
    const float* Wd = (const float*)d_in[1];
    const float* bd = (const float*)d_in[2];
    const float* Wu = (const float*)d_in[3];
    const float* bu = (const float*)d_in[4];
    const float* Wr = (const float*)d_in[5];
    const float* br = (const float*)d_in[6];
    float* out = (float*)d_out;

    char* ws = (char*)d_ws;
    _Float16* WdT = (_Float16*)(ws + WS_WDT);
    _Float16* WuT = (_Float16*)(ws + WS_WUT);
    float* partial = (float*)(ws + WS_WDT);   // reused before weight conversion
    float* pool    = (float*)(ws + WS_POOL);
    float* wbuf    = (float*)(ws + WS_W);
    float* bm      = (float*)(ws + WS_BM);

    pool_partial_k<<<dim3(8, 3, 16), 256, 0, stream>>>(x, partial);
    pool_reduce_k<<<dim3(8, 3), 256, 0, stream>>>(partial, pool);
    router_k<<<1, 32, 0, stream>>>(pool, Wr, br, wbuf);
    biasmix_k<<<1, 256, 0, stream>>>(wbuf, bu, bm);
    conv_wd_k<<<576, 256, 0, stream>>>(Wd, WdT);
    conv_wu_k<<<576, 256, 0, stream>>>(Wu, WuT);
    adapter_main_k<<<256, 256, 0, stream>>>(x, wbuf, bd, bm, WdT, WuT, out);
}
